// DirectConv2d_70712341562038
// MI455X (gfx1250) — 8-bit matrix operand order repaired and checked by reading, compiled, not run on hardware
//
#include <hip/hip_runtime.h>

#define BB    32
#define CIN   128
#define COUT  256
#define KK    3
#define HH    56
#define WW    60
#define PAD   2
#define OH    58
#define OW    62
#define PH    60
#define PW    68

typedef int v8i __attribute__((ext_vector_type(8)));
typedef int v4i __attribute__((ext_vector_type(4)));

#define PACKA_DWORDS (9 * 2 * 16 * 32 * 8)

__global__ void pack_weights_i8(const float* __restrict__ wf,
                                unsigned int* __restrict__ packedA) {
  int idx = blockIdx.x * blockDim.x + threadIdx.x;
  if (idx >= PACKA_DWORDS) return;
  int v       = idx & 7;
  int lane    = (idx >> 3) & 31;
  int co_tile = (idx >> 8) & 15;
  int chunk   = (idx >> 12) & 1;
  int tap     = idx >> 13;
  int kh = tap / 3, kw = tap % 3;
  int m     = lane & 15;
  int khalf = lane >> 4;
  int co    = co_tile * 16 + m;
  int kbase = (v >> 2) * 32 + khalf * 16 + (v & 3) * 4;
  unsigned int d = 0;
  #pragma unroll
  for (int j = 0; j < 4; ++j) {
    int ci = chunk * 64 + kbase + j;
    int w  = (int)wf[((size_t)(co * CIN + ci) * KK + kh) * KK + kw];
    d |= ((unsigned int)(w & 0xFF)) << (8 * j);
  }
  packedA[idx] = d;
}

#define PACKX_DWORDS (BB * PH * PW * (CIN / 4))

__global__ void pack_acts_i8(const float* __restrict__ xf,
                             unsigned int* __restrict__ p8) {
  int idx = blockIdx.x * blockDim.x + threadIdx.x;
  if (idx >= PACKX_DWORDS) return;
  int ci4  = idx & 31;
  int x    = (idx >> 5) % PW;
  int rest = (idx >> 5) / PW;
  int y    = rest % PH;
  int b    = rest / PH;
  int h = y - PAD, w = x - PAD;
  unsigned int d = 0;
  if (h >= 0 && h < HH && w >= 0 && w < WW) {
    #pragma unroll
    for (int j = 0; j < 4; ++j) {
      int ci = ci4 * 4 + j;
      int a  = (int)__builtin_nontemporal_load(
                   &xf[((size_t)(b * CIN + ci) * HH + h) * WW + w]);
      d |= ((unsigned int)(a & 0xFF)) << (8 * j);
    }
  }
  p8[idx] = d;
}

__device__ inline v8i load_ab(const unsigned char* p, int hi_off) {
  v4i lo = *(const v4i*)(p);
  v4i hi = *(const v4i*)(p + hi_off);
  v8i r;
  r[0] = lo[0]; r[1] = lo[1]; r[2] = lo[2]; r[3] = lo[3];
  r[4] = hi[0]; r[5] = hi[1]; r[6] = hi[2]; r[7] = hi[3];
  return r;
}

__global__ __launch_bounds__(64)
void conv2d_full_wmma_iu8(const unsigned char* __restrict__ p8,
                          const unsigned char* __restrict__ packedA,
                          float* __restrict__ out) {
  const int lane  = threadIdx.x & 31;
  const int wave  = threadIdx.x >> 5;
  const int cop   = blockIdx.x;
  const int oh    = blockIdx.y;
  const int b     = blockIdx.z;
  const int n     = lane & 15;
  const int khalf = lane >> 4;
  const int ow0   = wave * 32 + n;
  const int ct0   = cop * 2;

  const unsigned char* abase = packedA + (size_t)ct0 * 1024 + lane * 32;
  const unsigned char* bbase =
      p8 + (size_t)((b * PH + oh) * PW + ow0) * CIN + khalf * 16;

  v8i c00 = {};
  v8i c01 = {};
  v8i c10 = {};
  v8i c11 = {};

  #pragma unroll
  for (int tap = 0; tap < 9; ++tap) {
    const int kh = tap / 3, kw = tap % 3;
    const int boff_tap = (kh * PW + kw) * CIN;
    #pragma unroll
    for (int chunk = 0; chunk < 2; ++chunk) {
      const int aoff = (tap * 2 + chunk) * 16 * 1024;
      const int boff = boff_tap + chunk * 64;

      v8i a0 = load_ab(abase + aoff, 16);
      v8i a1 = load_ab(abase + aoff + 1024, 16);
      v8i b0 = load_ab(bbase + boff, 32);
      v8i b1 = load_ab(bbase + boff + 16 * CIN, 32);

      c00 = __builtin_amdgcn_wmma_i32_16x16x64_iu8(false, a0, false, b0, c00,
                                                   false, false);
      c01 = __builtin_amdgcn_wmma_i32_16x16x64_iu8(false, a0, false, b1, c01,
                                                   false, false);
      c10 = __builtin_amdgcn_wmma_i32_16x16x64_iu8(false, a1, false, b0, c10,
                                                   false, false);
      c11 = __builtin_amdgcn_wmma_i32_16x16x64_iu8(false, a1, false, b1, c11,
                                                   false, false);
    }
  }

  const int m0 = khalf * 8;
  const size_t cstride = (size_t)OH * OW;
  float* o00 = out + ((size_t)(b * COUT + ct0 * 16 + m0) * OH + oh) * OW + ow0;
  float* o10 = o00 + 16 * cstride;
  const bool ow1_ok = (ow0 + 16) < OW;

  #pragma unroll
  for (int v = 0; v < 8; ++v) {
    __builtin_nontemporal_store((float)c00[v], o00 + v * cstride);
    __builtin_nontemporal_store((float)c10[v], o10 + v * cstride);
    if (ow1_ok) {
      __builtin_nontemporal_store((float)c01[v], o00 + v * cstride + 16);
      __builtin_nontemporal_store((float)c11[v], o10 + v * cstride + 16);
    }
  }
}

extern "C" void kernel_launch(void* const* d_in, const int* in_sizes, int n_in,
                              void* d_out, int out_size, void* d_ws,
                              size_t ws_size, hipStream_t stream) {
  const float* inp = (const float*)d_in[0];
  const float* wgt = (const float*)d_in[1];
  float* out = (float*)d_out;

  unsigned char* packedA = (unsigned char*)d_ws;
  unsigned char* p8      = packedA + (1 << 19);

  pack_weights_i8<<<(PACKA_DWORDS + 255) / 256, 256, 0, stream>>>(
      wgt, (unsigned int*)packedA);
  pack_acts_i8<<<(PACKX_DWORDS + 255) / 256, 256, 0, stream>>>(
      inp, (unsigned int*)p8);

  dim3 grid(COUT / 32, OH, BB);
  conv2d_full_wmma_iu8<<<grid, 64, 0, stream>>>(p8, packedA, out);
}
